// Attention_layer_89721866813946
// MI455X (gfx1250) — matrix operand order repaired and checked by reading, not compiled, not run on hardware
//
#include <hip/hip_runtime.h>
#include <hip/hip_bf16.h>

// ---------------------------------------------------------------------------
// Sizes (fixed by the reference)
// ---------------------------------------------------------------------------
#define BB   2048
#define TT   200
#define DD   64
#define HH1  128
#define HH2  64
#define NTILE 13               // ceil(200/16)
#define NEG_INF (-4294967295.0f)   // -(2^32)+1

typedef __attribute__((ext_vector_type(16))) __bf16 v16bf;
typedef __attribute__((ext_vector_type(8)))  __bf16 v8bf;
typedef __attribute__((ext_vector_type(2)))  __bf16 v2bf;
typedef __attribute__((ext_vector_type(8)))  float  v8f;
typedef __attribute__((ext_vector_type(4)))  unsigned u32x4;
typedef __attribute__((ext_vector_type(8)))  unsigned u32x8;

union Frag16 { v16bf v; v8bf h[2]; };

// fp32 -> bf16 bits, round-to-nearest-even (scalar, used by prep kernel)
__device__ __forceinline__ unsigned short f2bf(float f) {
    union { float f; unsigned u; } x; x.f = f;
    unsigned u = x.u;
    u += 0x7FFFu + ((u >> 16) & 1u);
    return (unsigned short)(u >> 16);
}

// pack two fp32 -> {lo: bf16(a), hi: bf16(b)} in one dword
__device__ __forceinline__ unsigned pack_bf16(float a, float b) {
#if __has_builtin(__builtin_amdgcn_cvt_pk_bf16_f32)
    union { v2bf v; unsigned u; } r;
    r.v = __builtin_amdgcn_cvt_pk_bf16_f32(a, b);   // v_cvt_pk_bf16_f32 (HW RNE)
    return r.u;
#else
    union { float f; unsigned u; } x, y; x.f = a; y.f = b;
    return (y.u & 0xFFFF0000u) | (x.u >> 16);       // truncating pack
#endif
}

// low 32 bits of a generic pointer to a __shared__ object == LDS byte offset
__device__ __forceinline__ unsigned lds_off(const void* p) {
    return (unsigned)(unsigned long long)p;
}

// ---------------------------------------------------------------------------
// TDM: async DMA of a [rows<=16, 64] f32 tile (row stride 64) into LDS.
// D# per CDNA5 ISA:
//   group0: [1:0]=count=1, [63:32]=lds_addr, [120:64]=global_addr, [127:126]=2
//   group1: data_size=2(4B), tensor_dim0=64, tensor_dim1=rows (OOB rows -> 0),
//           tile_dim0=64, tile_dim1=16, tensor_dim0_stride=64
// Issued once per workgroup (wave 0); tracked by TENSORcnt.
// ---------------------------------------------------------------------------
__device__ __forceinline__ void tdm_load_key_tile(const float* gptr,
                                                  unsigned ldsaddr,
                                                  unsigned rows) {
    unsigned long long ga = (unsigned long long)gptr;
    u32x4 g0;
    g0[0] = 1u;                                            // count=1 (valid)
    g0[1] = ldsaddr;                                       // lds_addr
    g0[2] = (unsigned)ga;                                  // global_addr[31:0]
    g0[3] = ((unsigned)(ga >> 32) & 0x01FFFFFFu) | (2u << 30); // ga[56:32] | type=2
    u32x8 g1;
    g1[0] = 0x00020000u;          // workgroup_mask=0, data_size=2 (4 bytes)
    g1[1] = 64u << 16;            // tensor_dim0[15:0]=64  (bits[63:48])
    g1[2] = rows << 16;           // tensor_dim0 hi=0, tensor_dim1[15:0]=rows
    g1[3] = 64u << 16;            // tensor_dim1 hi=0, tile_dim0=64
    g1[4] = 16u;                  // tile_dim1=16, tile_dim2=0
    g1[5] = 64u;                  // tensor_dim0_stride = 64 (low 32)
    g1[6] = 0u;                   // stride0 hi, stride1 lo
    g1[7] = 0u;                   // stride1 hi (unused for 2D)
    u32x4 gz = {0u, 0u, 0u, 0u};  // groups 2/3: tensor_dim2=0, tile_dim3=0
    asm volatile("tensor_load_to_lds %0, %1, %2, %3"
                 :
                 : "s"(g0), "s"(g1), "s"(gz), "s"(gz)
                 : "memory");
}

// ---------------------------------------------------------------------------
// Prep: swizzle W1 [256,128] and W2 [128,64] (fp32, row-major) into bf16
// wave32 B-fragment layout:
//   fragment (nt, kt): 32 lanes x 16 bf16 contiguous per lane
//   lane L: column N = nt*16 + (L%16); K = kt*32 + (L/16)*16 + j, j=0..15
// ---------------------------------------------------------------------------
__global__ __launch_bounds__(256) void prep_weights(
    const float* __restrict__ W1, const float* __restrict__ W2,
    unsigned short* __restrict__ ws)
{
    int idx = blockIdx.x * 256 + threadIdx.x;
    if (idx < 64 * 32 * 16) {                       // W1 fragments
        int j    = idx & 15;
        int lane = (idx >> 4) & 31;
        int kt   = (idx >> 9) & 7;
        int nt   = idx >> 12;
        int K    = kt * 32 + (lane >> 4) * 16 + j;
        int col  = nt * 16 + (lane & 15);
        ws[idx]  = f2bf(W1[K * HH1 + col]);
    } else if (idx < 64 * 32 * 16 + 16 * 32 * 16) { // W2 fragments
        int r    = idx - 32768;
        int j    = r & 15;
        int lane = (r >> 4) & 31;
        int kt   = (r >> 9) & 3;
        int nt   = r >> 11;
        int K    = kt * 32 + (lane >> 4) * 16 + j;
        int col  = nt * 16 + (lane & 15);
        ws[idx]  = f2bf(W2[K * HH2 + col]);
    }
}

// ---------------------------------------------------------------------------
// Main: one workgroup (8 waves) per batch element.
// ---------------------------------------------------------------------------
__global__ __launch_bounds__(256) void attn_main(
    const float* __restrict__ query, const float* __restrict__ key,
    const float* __restrict__ value, const int* __restrict__ mask,
    const float* __restrict__ b1,    const float* __restrict__ b2,
    const float* __restrict__ Wo,    const float* __restrict__ bo,
    const unsigned short* __restrict__ W1f,
    const unsigned short* __restrict__ W2f,
    float* __restrict__ out)
{
    __shared__ __align__(16) float qs[DD];
    __shared__ __align__(16) float kstage[2][16][DD];       // TDM double buffer (8 KB)
    __shared__ __align__(16) unsigned short emb[16][256];   // 8 KB bf16
    __shared__ __align__(16) unsigned short h1s[16][HH1];   // 4 KB bf16
    __shared__ __align__(16) float sc[NTILE * 16];          // scores (208)
    __shared__ __align__(16) float red[256];
    __shared__ __align__(16) float pv[4][DD];

    const int tid  = threadIdx.x;
    const int w    = tid >> 5;        // wave id 0..7
    const int lane = tid & 31;
    const int half = lane >> 4;       // 0: lanes 0-15, 1: lanes 16-31
    const int l16  = lane & 15;
    const int b    = blockIdx.x;
    const float* keyb = key + (size_t)b * TT * DD;
    const float* vb   = value + (size_t)b * TT * DD;

    // ---- register-resident B fragments (reused for all 13 row tiles) ----
    Frag16 w1frag[8];
#pragma unroll
    for (int kt = 0; kt < 8; ++kt) {
        const unsigned short* p = W1f + (((size_t)(w * 8 + kt) * 32 + lane) * 16);
        w1frag[kt].h[0] = *(const v8bf*)p;
        w1frag[kt].h[1] = *(const v8bf*)(p + 8);
    }
    Frag16 w2frag[4];
    if (w < 4) {
#pragma unroll
        for (int kt = 0; kt < 4; ++kt) {
            const unsigned short* p = W2f + (((size_t)(w * 4 + kt) * 32 + lane) * 16);
            w2frag[kt].h[0] = *(const v8bf*)p;
            w2frag[kt].h[1] = *(const v8bf*)(p + 8);
        }
    }

    const float b1v = b1[w * 16 + l16];
    const float b2v = (w < 4) ? b2[w * 16 + l16] : 0.f;
    const float wov = (w < 4) ? Wo[w * 16 + l16] : 0.f;
    const float bo0 = bo[0];
    const int   col = w * 16 + l16;   // h1 column owned by this lane

    if (tid < DD) qs[tid] = query[(size_t)b * DD + tid];

    // prologue: DMA key tile 0 into buffer 0
    if (w == 0) {
        tdm_load_key_tile(keyb, lds_off(&kstage[0][0][0]), 16u);
        __builtin_amdgcn_s_wait_tensorcnt(0);
    }
    __syncthreads();

    // =====================  row tiles of 16  ============================
    for (int tt = 0; tt < NTILE; ++tt) {
        const int t0  = tt * 16;
        const int buf = tt & 1;

        // kick off async DMA of the NEXT key tile (overlaps GEMM1+GEMM2)
        if (w == 0 && tt + 1 < NTILE) {
            const int t1 = t0 + 16;
            unsigned rows = (unsigned)((TT - t1 < 16) ? (TT - t1) : 16);
            tdm_load_key_tile(keyb + (size_t)t1 * DD,
                              lds_off(&kstage[buf ^ 1][0][0]), rows);
        }

        // ---- build emb tile [16,256] bf16: [q | k | q-k | q*k] ----
        // each thread owns column pairs -> packed b32 LDS stores
        {
            const int row   = tid >> 4;
            const int cbase = (tid & 15) * 2;
#pragma unroll
            for (int cc = 0; cc < DD; cc += 32) {
                const int c = cbase + cc;
                float k0 = kstage[buf][row][c];       // OOB rows zero-filled by TDM
                float k1 = kstage[buf][row][c + 1];
                float q0 = qs[c];
                float q1 = qs[c + 1];
                *(unsigned*)&emb[row][c]        = pack_bf16(q0, q1);
                *(unsigned*)&emb[row][64 + c]   = pack_bf16(k0, k1);
                *(unsigned*)&emb[row][128 + c]  = pack_bf16(q0 - k0, q1 - k1);
                *(unsigned*)&emb[row][192 + c]  = pack_bf16(q0 * k0, q1 * k1);
            }
            if (tid < 16) sc[t0 + tid] = bo0;   // init score with bias
        }
        __syncthreads();

        // ---- GEMM1: [16,256] x [256, 16 (this wave's N-tile)] ----
        // explicit A-fragment double buffer: loads for kt+1 fly under WMMA kt
        v8f c1 = {};
        {
            Frag16 afr[2];
            const unsigned short* p0 = &emb[l16][half * 16];
            afr[0].h[0] = *(const v8bf*)p0;
            afr[0].h[1] = *(const v8bf*)(p0 + 8);
#pragma unroll
            for (int kt = 0; kt < 8; ++kt) {
                if (kt + 1 < 8) {
                    const unsigned short* p = &emb[l16][(kt + 1) * 32 + half * 16];
                    afr[(kt + 1) & 1].h[0] = *(const v8bf*)p;
                    afr[(kt + 1) & 1].h[1] = *(const v8bf*)(p + 8);
                }
                c1 = __builtin_amdgcn_wmma_f32_16x16x32_bf16(
                         false, afr[kt & 1].v, false, w1frag[kt].v,
                         (short)0, c1, false, false);
            }
        }
        // bias + relu + packed bf16 store of h1 tile (row pairs share a dword)
#pragma unroll
        for (int i = 0; i < 8; i += 2) {
            float h0 = fmaxf(c1[i] + b1v, 0.f);
            float h1 = fmaxf(c1[i + 1] + b1v, 0.f);
            unsigned pk = pack_bf16(h0, h1);
            h1s[half * 8 + i][col]     = (unsigned short)pk;         // ds_store_b16
            h1s[half * 8 + i + 1][col] = (unsigned short)(pk >> 16); // ds_store_b16_d16_hi
        }
        __syncthreads();

        // ---- GEMM2 + Wo dot (waves 0..3); waves 4..7 prefetch value ----
        if (w < 4) {
            v8f c2 = {};
            Frag16 afr[2];
            const unsigned short* p0 = &h1s[l16][half * 16];
            afr[0].h[0] = *(const v8bf*)p0;
            afr[0].h[1] = *(const v8bf*)(p0 + 8);
#pragma unroll
            for (int kt = 0; kt < 4; ++kt) {
                if (kt + 1 < 4) {
                    const unsigned short* p = &h1s[l16][(kt + 1) * 32 + half * 16];
                    afr[(kt + 1) & 1].h[0] = *(const v8bf*)p;
                    afr[(kt + 1) & 1].h[1] = *(const v8bf*)(p + 8);
                }
                c2 = __builtin_amdgcn_wmma_f32_16x16x32_bf16(
                         false, afr[kt & 1].v, false, w2frag[kt].v,
                         (short)0, c2, false, false);
            }
#pragma unroll
            for (int i = 0; i < 8; ++i) {
                float hv = fmaxf(c2[i] + b2v, 0.f);
                float p = hv * wov;                    // partial of row M over this lane's column
#pragma unroll
                for (int off = 1; off < 16; off <<= 1) // reduce 16 lanes (halves stay separate)
                    p += __shfl_xor(p, off, 32);
                if (l16 == 0)
                    atomicAdd(&sc[t0 + half * 8 + i], p);  // ds_add_f32, 4 waves combine
            }
        } else {
            // prefetch this tile's slice of `value` for the output phase
            const int tid2 = tid - 128;                // 0..127
            const int eoff = t0 * DD + tid2 * 32;      // 32-float stride covers 16KB
            if (eoff < TT * DD)
                __builtin_prefetch(vb + eoff, 0, 3);   // global_prefetch_b8
        }

        if (w == 0) __builtin_amdgcn_s_wait_tensorcnt(0);  // next key tile landed
        __syncthreads();
    }

    // =====================  masked softmax over T  ======================
    float s = -3.4e38f;
    if (tid < TT) {
        s = (mask[(size_t)b * TT + tid] == 0) ? NEG_INF : sc[tid];
        sc[tid] = s;
    }
    red[tid] = s;
    __syncthreads();
#pragma unroll
    for (int st = 128; st > 0; st >>= 1) {
        if (tid < st) red[tid] = fmaxf(red[tid], red[tid + st]);
        __syncthreads();
    }
    const float mx = red[0];
    __syncthreads();

    float p = 0.f;
    if (tid < TT) {
        p = __expf(sc[tid] - mx);   // masked entries underflow to 0
        sc[tid] = p;
    }
    red[tid] = p;
    __syncthreads();
#pragma unroll
    for (int st = 128; st > 0; st >>= 1) {
        if (tid < st) red[tid] += red[tid + st];
        __syncthreads();
    }
    const float inv = 1.0f / red[0];
    __syncthreads();

    // =====================  out = (sum_t p_t * v_t) * inv  ==============
    {
        const int g = tid >> 6;        // 0..3
        const int d = tid & 63;
        float acc = 0.f;
        for (int t = g; t < TT; t += 4)
            acc += sc[t] * vb[(size_t)t * DD + d];
        pv[g][d] = acc;
    }
    __syncthreads();
    if (tid < DD)
        out[(size_t)b * DD + tid] =
            (pv[0][tid] + pv[1][tid] + pv[2][tid] + pv[3][tid]) * inv;
}

// ---------------------------------------------------------------------------
extern "C" void kernel_launch(void* const* d_in, const int* in_sizes, int n_in,
                              void* d_out, int out_size, void* d_ws, size_t ws_size,
                              hipStream_t stream) {
    const float* query = (const float*)d_in[0];
    const float* key   = (const float*)d_in[1];
    const float* value = (const float*)d_in[2];
    const int*   mask  = (const int*)  d_in[3];
    const float* W1    = (const float*)d_in[4];
    const float* b1    = (const float*)d_in[5];
    const float* W2    = (const float*)d_in[6];
    const float* b2    = (const float*)d_in[7];
    const float* Wo    = (const float*)d_in[8];
    const float* bo    = (const float*)d_in[9];

    unsigned short* wfrag = (unsigned short*)d_ws;   // 64KB W1 frags + 16KB W2 frags

    prep_weights<<<160, 256, 0, stream>>>(W1, W2, wfrag);
    attn_main<<<BB, 256, 0, stream>>>(query, key, value, mask,
                                      b1, b2, Wo, bo,
                                      wfrag, wfrag + 32768,
                                      (float*)d_out);
    (void)in_sizes; (void)n_in; (void)out_size; (void)ws_size;
}
